// LocalEncoder_80272938762634
// MI455X (gfx1250) — matrix operand order repaired and checked by reading, not compiled, not run on hardware
//
#include <hip/hip_runtime.h>

#define WINSZ   16
#define HEADS   16
#define DMODEL  1024
#define NORIG   8192
#define NLAYERS 4

typedef __attribute__((ext_vector_type(16))) __bf16 v16bf;
typedef __attribute__((ext_vector_type(8)))  float  v8f;
typedef __attribute__((ext_vector_type(4)))  unsigned int u32x4;
typedef __attribute__((ext_vector_type(8)))  int i32x8;
typedef __attribute__((ext_vector_type(4)))  int i32x4;

// ---------------------------------------------------------------------------
// TDM: load a [tile_rows x tile_k] bf16 tile (K contiguous, row pitch k_len)
// into LDS with 16B pad after each 64B row (LDS row pitch = 80B).
// Descriptor packing per CDNA5 ISA.
// This toolchain exposes the 6-arg builtin: (g0, g1, g2, g3, g_extra, cpol).
// ---------------------------------------------------------------------------
#if __has_builtin(__builtin_amdgcn_tensor_load_to_lds)
#define USE_TDM 1
__device__ __forceinline__ void tdm_load_tile(const void* gbase, unsigned ldsoff,
                                              unsigned rows_total, unsigned k_len,
                                              unsigned row0, unsigned k0) {
  unsigned long long ga = (unsigned long long)(size_t)gbase
                        + 2ull * ((unsigned long long)row0 * k_len + k0);
  u32x4 g0 = { 1u,                                   // count=1 valid descriptor
               ldsoff,                               // lds_addr
               (unsigned)(ga & 0xffffffffu),
               ((unsigned)(ga >> 32) & 0x01ffffffu) | (2u << 30) };  // type=2
  // data_size=1 (2B), pad_enable, pad_interval=3 (16 dwords = 64B),
  // pad_amount=3 (4 dwords = 16B)
  unsigned w0 = (1u << 16) | (1u << 20) | (3u << 22) | (3u << 25);
  unsigned w1 = (k_len & 0xffffu) << 16;                       // tensor_dim0 lo
  unsigned w2 = ((k_len >> 16) & 0xffffu) | ((rows_total & 0xffffu) << 16);
  unsigned w3 = ((rows_total >> 16) & 0xffffu) | (32u << 16);  // tile_dim0 = 32
  unsigned w4 = 128u;                                          // tile_dim1 = 128
  unsigned w5 = k_len;                                         // dim0_stride
  i32x8 g1 = { (int)w0, (int)w1, (int)w2, (int)w3, (int)w4, (int)w5, 0, 0 };
  i32x4 z4 = { 0, 0, 0, 0 };
  i32x8 z8 = { 0, 0, 0, 0, 0, 0, 0, 0 };
  __builtin_amdgcn_tensor_load_to_lds(g0, g1, z4, z4, z8, 0);
}
#else
#define USE_TDM 0
#endif

// ---------------------------------------------------------------------------
// GEMM: C[M,N] = A[M,1024](bf16) * Wt[N,1024]^T(bf16) + bias (+resid)
// block = 256 thr (8 waves), tile 128x128, wave tile 64x32 (4x2 WMMA tiles)
// ---------------------------------------------------------------------------
__global__ __launch_bounds__(256)
void k_gemm_bf16(const __bf16* __restrict__ A, const __bf16* __restrict__ Bt,
                 const float* __restrict__ bias, const float* __restrict__ resid,
                 float* __restrict__ C, int M, int N)
{
  constexpr int K = 1024;
  constexpr int PITCH = 80;                       // 64B data + 16B pad
  __shared__ __align__(16) unsigned char ldsA[2][128 * PITCH];
  __shared__ __align__(16) unsigned char ldsB[2][128 * PITCH];

  const int tid  = threadIdx.x;
  const int lane = tid & 31;
  const int hx   = lane >> 4;                     // half of the wave
  const int l16  = lane & 15;
  const int wid  = __builtin_amdgcn_readfirstlane((int)tid) >> 5;  // uniform
  const int wm   = wid & 1;
  const int wn   = wid >> 1;
  const int m0   = blockIdx.x * 128;
  const int n0   = blockIdx.y * 128;

  __builtin_prefetch(bias + n0, 0, 0);            // global_prefetch_b8

  v8f zero8 = {0.f,0.f,0.f,0.f,0.f,0.f,0.f,0.f};
  v8f acc[4][2];
  for (int i = 0; i < 4; ++i)
    for (int j = 0; j < 2; ++j) acc[i][j] = zero8;

  const int nk = K / 32;
#if USE_TDM
  if (wid == 0) {
    tdm_load_tile(A,  (unsigned)(size_t)&ldsA[0][0], (unsigned)M, K, m0, 0);
    tdm_load_tile(Bt, (unsigned)(size_t)&ldsB[0][0], (unsigned)N, K, n0, 0);
  }
#endif
  for (int ks = 0; ks < nk; ++ks) {
    const int buf = ks & 1;
#if USE_TDM
    if (wid == 0) {
      if (ks + 1 < nk) {
        tdm_load_tile(A,  (unsigned)(size_t)&ldsA[buf ^ 1][0], (unsigned)M, K, m0, (ks + 1) * 32);
        tdm_load_tile(Bt, (unsigned)(size_t)&ldsB[buf ^ 1][0], (unsigned)N, K, n0, (ks + 1) * 32);
        __builtin_amdgcn_s_wait_tensorcnt(2);     // current tile's 2 DMAs done
      } else {
        __builtin_amdgcn_s_wait_tensorcnt(0);
      }
    }
#else
    // fallback: cooperative global->LDS copy, same padded layout
    {
      const int k0 = ks * 32;
      for (int i = 0; i < 2; ++i) {
        int chunk = tid + 256 * i;                // 512 chunks of 16B per tile
        int row = chunk >> 2, seg = chunk & 3;
        uint4 va = {0u, 0u, 0u, 0u};
        if (m0 + row < M)
          va = *(const uint4*)(A + (size_t)(m0 + row) * K + k0 + seg * 8);
        *(uint4*)(&ldsA[buf][row * PITCH + seg * 16]) = va;
        uint4 vb = *(const uint4*)(Bt + (size_t)(n0 + row) * K + k0 + seg * 8);
        *(uint4*)(&ldsB[buf][row * PITCH + seg * 16]) = vb;
      }
    }
#endif
    __syncthreads();

    v16bf afrag[4], bfrag[2];
#pragma unroll
    for (int mt = 0; mt < 4; ++mt) {              // A 16x32: K in {8h..,16+8h..}
      int row = wm * 64 + mt * 16 + l16;
      const uint4* p = (const uint4*)(&ldsA[buf][row * PITCH + 16 * hx]);
      union { uint4 q[2]; v16bf v; } u;
      u.q[0] = p[0]; u.q[1] = p[2];
      afrag[mt] = u.v;
    }
#pragma unroll
    for (int nt = 0; nt < 2; ++nt) {              // B 32x16: K in {8h..,16+8h..}
      int row = wn * 32 + nt * 16 + l16;
      const uint4* p = (const uint4*)(&ldsB[buf][row * PITCH + 16 * hx]);
      union { uint4 q[2]; v16bf v; } u;
      u.q[0] = p[0]; u.q[1] = p[2];
      bfrag[nt] = u.v;
    }
#pragma unroll
    for (int mt = 0; mt < 4; ++mt)
#pragma unroll
      for (int nt = 0; nt < 2; ++nt)
        acc[mt][nt] = __builtin_amdgcn_wmma_f32_16x16x32_bf16(
            false, afrag[mt], false, bfrag[nt], (short)0, acc[mt][nt], false, false);
    __syncthreads();
  }

#pragma unroll
  for (int mt = 0; mt < 4; ++mt)
#pragma unroll
    for (int nt = 0; nt < 2; ++nt) {
      int col = n0 + wn * 32 + nt * 16 + l16;
      float bv = bias[col];
#pragma unroll
      for (int v = 0; v < 8; ++v) {
        int row = m0 + wm * 64 + mt * 16 + v + 8 * hx;
        if (row < M) {
          float val = acc[mt][nt][v] + bv;
          if (resid) val += resid[(size_t)row * N + col];
          C[(size_t)row * N + col] = val;
        }
      }
    }
}

// ---------------------------------------------------------------------------
// LayerNorm -> bf16. One wave per row (1024 cols), 8 rows per block.
// ---------------------------------------------------------------------------
__global__ __launch_bounds__(256)
void k_layernorm_bf16(const float* __restrict__ x, const float* __restrict__ g,
                      const float* __restrict__ b, __bf16* __restrict__ h, int M)
{
  const int lane = threadIdx.x & 31;
  const int row  = blockIdx.x * 8 + (threadIdx.x >> 5);
  if (row >= M) return;
  const float* xr = x + (size_t)row * DMODEL;
  float vals[32];
  float s = 0.f;
#pragma unroll
  for (int i = 0; i < 32; ++i) { vals[i] = xr[lane + 32 * i]; s += vals[i]; }
#pragma unroll
  for (int m = 16; m >= 1; m >>= 1) s += __shfl_xor(s, m, 32);
  float mean = s * (1.0f / DMODEL);
  float var = 0.f;
#pragma unroll
  for (int i = 0; i < 32; ++i) { float d = vals[i] - mean; var += d * d; }
#pragma unroll
  for (int m = 16; m >= 1; m >>= 1) var += __shfl_xor(var, m, 32);
  float rs = rsqrtf(var * (1.0f / DMODEL) + 1e-5f);
  __bf16* hr = h + (size_t)row * DMODEL;
#pragma unroll
  for (int i = 0; i < 32; ++i) {
    int c = lane + 32 * i;
    hr[c] = (__bf16)((vals[i] - mean) * rs * g[c] + b[c]);
  }
}

// ---------------------------------------------------------------------------
// Windowed attention: block = window (512 thr), wave = head.
// S = QK^T/8 (2x wmma K=32), softmax over the 16-lane half-groups,
// O = P V (4x wmma, K=16 zero-padded to 32). o written in bf16.
// ---------------------------------------------------------------------------
__global__ __launch_bounds__(512)
void k_attn(const float* __restrict__ qkv, __bf16* __restrict__ o, int nw)
{
  __shared__ float P[16][16][16];                 // [head][q][ktok]
  const int w    = blockIdx.x;
  const int lane = threadIdx.x & 31;
  const int head = threadIdx.x >> 5;
  const int hx   = lane >> 4;
  const int l16  = lane & 15;
  const size_t rowbase = (size_t)w * WINSZ;
  if (w >= nw) return;

  v8f s = {0.f,0.f,0.f,0.f,0.f,0.f,0.f,0.f};
#pragma unroll
  for (int c = 0; c < 2; ++c) {
    __align__(16) float a16[16];
    const float* qrow = qkv + (rowbase + l16) * 3072 + head * 64 + c * 32;
    *(float4*)(a16 + 0)  = *(const float4*)(qrow + 8 * hx);
    *(float4*)(a16 + 4)  = *(const float4*)(qrow + 8 * hx + 4);
    *(float4*)(a16 + 8)  = *(const float4*)(qrow + 16 + 8 * hx);
    *(float4*)(a16 + 12) = *(const float4*)(qrow + 16 + 8 * hx + 4);
    union { v16bf v; __bf16 e[16]; } Af, Bf;
#pragma unroll
    for (int e = 0; e < 16; ++e) Af.e[e] = (__bf16)a16[e];
    __align__(16) float b16[16];
    const float* krow = qkv + (rowbase + l16) * 3072 + 1024 + head * 64 + c * 32 + 8 * hx;
    *(float4*)(b16 + 0)  = *(const float4*)(krow + 0);
    *(float4*)(b16 + 4)  = *(const float4*)(krow + 4);
    *(float4*)(b16 + 8)  = *(const float4*)(krow + 16);
    *(float4*)(b16 + 12) = *(const float4*)(krow + 20);
#pragma unroll
    for (int e = 0; e < 16; ++e) Bf.e[e] = (__bf16)b16[e];
    s = __builtin_amdgcn_wmma_f32_16x16x32_bf16(false, Af.v, false, Bf.v,
                                                (short)0, s, false, false);
  }

  float pv[8];
#pragma unroll
  for (int v = 0; v < 8; ++v) {                   // row = v + 8*hx, col = l16
    float t = s[v] * 0.125f;
    float mx = t;
#pragma unroll
    for (int m = 8; m >= 1; m >>= 1) mx = fmaxf(mx, __shfl_xor(mx, m, 32));
    float e = __expf(t - mx);
    float sum = e;
#pragma unroll
    for (int m = 8; m >= 1; m >>= 1) sum += __shfl_xor(sum, m, 32);
    pv[v] = e / sum;
  }
#pragma unroll
  for (int v = 0; v < 8; ++v) P[head][v + 8 * hx][l16] = pv[v];
  __syncthreads();

  union { v16bf v; __bf16 e[16]; } Ap;
#pragma unroll
  for (int e = 0; e < 16; ++e) Ap.e[e] = (__bf16)(hx == 0 ? P[head][l16][e] : 0.0f);
#pragma unroll
  for (int e = 16; e < 16; ++e) Ap.e[e] = (__bf16)0.0f;   // K padding 16..31 (hx=1 half)

#pragma unroll
  for (int cc = 0; cc < 4; ++cc) {
    union { v16bf v; __bf16 e[16]; } Bv;
    if (hx == 0) {
      int d = cc * 16 + l16;
#pragma unroll
      for (int e = 0; e < 16; ++e)
        Bv.e[e] = (__bf16)qkv[(rowbase + e) * 3072 + 2048 + head * 64 + d];
    } else {
#pragma unroll
      for (int e = 0; e < 16; ++e) Bv.e[e] = (__bf16)0.0f;
    }
    v8f oacc = {0.f,0.f,0.f,0.f,0.f,0.f,0.f,0.f};
    oacc = __builtin_amdgcn_wmma_f32_16x16x32_bf16(false, Ap.v, false, Bv.v,
                                                   (short)0, oacc, false, false);
#pragma unroll
    for (int v = 0; v < 8; ++v)
      o[(rowbase + v + 8 * hx) * (size_t)DMODEL + head * 64 + cc * 16 + l16] =
          (__bf16)oacc[v];
  }
}

// ---------------------------------------------------------------------------
// ToMe merge: one block per window. Produces merged x and the old->new
// current-token remap (merge is a partition -> source tracked via assign[]).
// ---------------------------------------------------------------------------
__global__ __launch_bounds__(256)
void k_merge(const float* __restrict__ xin, float* __restrict__ xout,
             int* __restrict__ remap, int r)
{
  __shared__ float norms[16];
  __shared__ float sc[8][8];
  __shared__ int s_unm[8], s_src[8], s_dstj[8], s_cnt[8];
  const int w = blockIdx.x;
  const int t = threadIdx.x;
  const size_t base = (size_t)w * WINSZ * DMODEL;

  {                                               // row L2 norms
    int rr = t >> 4, ll = t & 15;
    float ss = 0.f;
    for (int c = ll; c < DMODEL; c += 16) {
      float v = xin[base + rr * DMODEL + c];
      ss += v * v;
    }
#pragma unroll
    for (int m = 8; m >= 1; m >>= 1) ss += __shfl_xor(ss, m, 32);
    if (ll == 0) norms[rr] = sqrtf(ss);
  }
  __syncthreads();
  if (t < 64) {                                   // cosine scores a_i . b_j
    int i = t >> 3, j = t & 7;
    const float* ar = xin + base + (2 * i) * DMODEL;
    const float* br = xin + base + (2 * j + 1) * DMODEL;
    float d = 0.f;
    for (int c = 0; c < DMODEL; ++c) d += ar[c] * br[c];
    sc[i][j] = d / ((norms[2 * i] + 1e-6f) * (norms[2 * j + 1] + 1e-6f));
  }
  __syncthreads();
  if (t == 0) {                                   // argmax + stable desc sort
    float nmax[8]; int nidx[8], order[8], cnt[8];
    for (int i = 0; i < 8; ++i) {
      float mx = sc[i][0]; int ix = 0;
      for (int j = 1; j < 8; ++j) if (sc[i][j] > mx) { mx = sc[i][j]; ix = j; }
      nmax[i] = mx; nidx[i] = ix; order[i] = i; cnt[i] = 0;
    }
    for (int i = 1; i < 8; ++i) {                 // stable insertion sort (desc)
      int oi = order[i]; float key = nmax[oi]; int j = i - 1;
      while (j >= 0 && nmax[order[j]] < key) { order[j + 1] = order[j]; --j; }
      order[j + 1] = oi;
    }
    int outw = WINSZ - r;
    for (int q = 0; q < r; ++q) {
      int s0 = order[q]; int dj = nidx[s0];
      s_src[q] = s0; s_dstj[q] = dj; cnt[dj]++;
      remap[w * WINSZ + 2 * s0] = w * outw + (8 - r) + dj;
    }
    for (int p = 0; p < 8 - r; ++p) {
      s_unm[p] = order[r + p];
      remap[w * WINSZ + 2 * order[r + p]] = w * outw + p;
    }
    for (int j = 0; j < 8; ++j) {
      s_cnt[j] = cnt[j];
      remap[w * WINSZ + 2 * j + 1] = w * outw + (8 - r) + j;
    }
  }
  __syncthreads();
  const int outw = WINSZ - r;
  for (int c = t; c < DMODEL; c += 256) {
    for (int p = 0; p < outw; ++p) {
      float val;
      if (p < 8 - r) {
        val = xin[base + (2 * s_unm[p]) * DMODEL + c];
      } else {
        int j = p - (8 - r);
        val = xin[base + (2 * j + 1) * DMODEL + c];
        for (int q = 0; q < r; ++q)
          if (s_dstj[q] == j) val += xin[base + (2 * s_src[q]) * DMODEL + c];
        val *= 1.0f / (1.0f + (float)s_cnt[j]);
      }
      xout[((size_t)w * outw + p) * DMODEL + c] = val;
    }
  }
}

// ---------------------------------------------------------------------------
// small helpers
// ---------------------------------------------------------------------------
__global__ void k_embed(const int* __restrict__ ids, const float* __restrict__ emb,
                        float* __restrict__ x) {
  size_t t = blockIdx.x * (size_t)blockDim.x + threadIdx.x;
  if (t < (size_t)NORIG * DMODEL) {
    int n = (int)(t >> 10), d = (int)(t & 1023);
    x[t] = emb[(size_t)ids[n] * DMODEL + d];
  }
}
__global__ void k_init_assign(int* a) {
  int t = blockIdx.x * blockDim.x + threadIdx.x;
  if (t < NORIG) a[t] = t;
}
__global__ void k_assign_update(int* __restrict__ a, const int* __restrict__ remap) {
  int t = blockIdx.x * blockDim.x + threadIdx.x;
  if (t < NORIG) a[t] = remap[a[t]];
}
__global__ void k_wt(const float* __restrict__ w, __bf16* __restrict__ wt,
                     int K, int N) {                  // w: KxN -> wt: NxK bf16
  size_t t = blockIdx.x * (size_t)blockDim.x + threadIdx.x;
  if (t < (size_t)K * N) {
    int n = (int)(t / K), k = (int)(t % K);
    wt[t] = (__bf16)w[(size_t)k * N + n];
  }
}
__global__ void k_copy(const float* __restrict__ s, float* __restrict__ d, size_t n) {
  size_t t = blockIdx.x * (size_t)blockDim.x + threadIdx.x;
  if (t < n) d[t] = s[t];
}
__global__ void k_source_out(const int* __restrict__ assign, float* __restrict__ out,
                             int curf) {               // (8192 x curf) one-hot
  size_t t = blockIdx.x * (size_t)blockDim.x + threadIdx.x;
  size_t total = (size_t)NORIG * (size_t)curf;
  if (t < total) {
    int row = (int)(t / curf), j = (int)(t % curf);
    out[t] = (assign[row] == j) ? 1.0f : 0.0f;
  }
}

// ---------------------------------------------------------------------------
extern "C" void kernel_launch(void* const* d_in, const int* in_sizes, int n_in,
                              void* d_out, int out_size, void* d_ws, size_t ws_size,
                              hipStream_t stream)
{
  const int*   ids  = (const int*)d_in[0];
  const float* emb  = (const float*)d_in[1];
  const float* Wqkv = (const float*)d_in[2];
  const float* bqkv = (const float*)d_in[3];
  const float* Wo   = (const float*)d_in[4];
  const float* bo   = (const float*)d_in[5];
  const float* lng  = (const float*)d_in[6];
  const float* lnb  = (const float*)d_in[7];
  float* out = (float*)d_out;

  char* ws = (char*)d_ws;
  size_t off = 0;
  auto alloc = [&](size_t bytes) -> void* {
    void* p = ws + off; off += (bytes + 255) & ~(size_t)255; return p;
  };
  float*  xA     = (float*)alloc((size_t)NORIG * DMODEL * 4);
  float*  xB     = (float*)alloc((size_t)NORIG * DMODEL * 4);
  __bf16* hbuf   = (__bf16*)alloc((size_t)NORIG * DMODEL * 2);
  float*  qkvbuf = (float*)alloc((size_t)NORIG * 3072 * 4);
  __bf16* obuf   = (__bf16*)alloc((size_t)NORIG * DMODEL * 2);
  __bf16* WqkvT  = (__bf16*)alloc((size_t)NLAYERS * 3072 * 1024 * 2);
  __bf16* WoT    = (__bf16*)alloc((size_t)NLAYERS * 1024 * 1024 * 2);
  int*    assign = (int*)alloc(NORIG * 4);
  int*    remap  = (int*)alloc(NORIG * 4);

  k_embed<<<(NORIG * DMODEL + 255) / 256, 256, 0, stream>>>(ids, emb, xA);
  k_init_assign<<<NORIG / 256, 256, 0, stream>>>(assign);
  for (int l = 0; l < NLAYERS; ++l) {
    k_wt<<<(1024 * 3072 + 255) / 256, 256, 0, stream>>>(
        Wqkv + (size_t)l * 1024 * 3072, WqkvT + (size_t)l * 3072 * 1024, 1024, 3072);
    k_wt<<<(1024 * 1024 + 255) / 256, 256, 0, stream>>>(
        Wo + (size_t)l * 1024 * 1024, WoT + (size_t)l * 1024 * 1024, 1024, 1024);
  }

  int cur = NORIG;
  const int rlist[NLAYERS] = {2, 2, 2, 4};
  for (int l = 0; l < NLAYERS; ++l) {
    const int r = rlist[l];
    const int nw = cur / WINSZ;
    k_layernorm_bf16<<<cur / 8, 256, 0, stream>>>(xA, lng + l * 1024, lnb + l * 1024,
                                                  hbuf, cur);
    dim3 g1((cur + 127) / 128, 3072 / 128);
    k_gemm_bf16<<<g1, 256, 0, stream>>>(hbuf, WqkvT + (size_t)l * 3072 * 1024,
                                        bqkv + l * 3072, nullptr, qkvbuf, cur, 3072);
    k_attn<<<nw, 512, 0, stream>>>(qkvbuf, obuf, nw);
    dim3 g2((cur + 127) / 128, 1024 / 128);
    k_gemm_bf16<<<g2, 256, 0, stream>>>(obuf, WoT + (size_t)l * 1024 * 1024,
                                        bo + l * 1024, xA, xB, cur, 1024);
    k_merge<<<nw, 256, 0, stream>>>(xB, xA, remap, r);
    k_assign_update<<<NORIG / 256, 256, 0, stream>>>(assign, remap);
    cur = nw * (WINSZ - r);
  }

  // outputs: x (cur x 1024) then source (8192 x cur) one-hot from assign[]
  k_copy<<<(unsigned)(((size_t)cur * DMODEL + 255) / 256), 256, 0, stream>>>(
      xA, out, (size_t)cur * DMODEL);
  size_t stotal = (size_t)NORIG * cur;
  k_source_out<<<(unsigned)((stotal + 255) / 256), 256, 0, stream>>>(
      assign, out + (size_t)cur * DMODEL, cur);
}
